// SubBlock_53772990546086
// MI455X (gfx1250) — matrix operand order repaired and checked by reading, not compiled, not run on hardware
//
#include <hip/hip_runtime.h>
#include <math.h>

// ============================================================================
// CDNA5 (gfx1250, wave32) implementation of the equivariant GNN block.
// All dense linear algebra runs through one generic WMMA GEMM kernel using
// v_wmma_f32_16x16x32_f16 (f16 in, f32 accumulate), templated on the
// TensLinear spatial factor (1/3/9), N-tiles-per-wave (1..4) and K-tail.
// Fragment loads are explicit float4 (global_load_b128) where alignment is
// guaranteed by construction.
// ============================================================================

typedef __attribute__((ext_vector_type(16))) _Float16 v16h;
typedef __attribute__((ext_vector_type(8)))  float    v8f;
typedef __attribute__((ext_vector_type(4)))  float    f4;

static inline int cdiv(int a, int b) { return (a + b - 1) / b; }

// ---------------------------------------------------------------------------
// Generic WMMA GEMM:
//   Y[(m/SPAT), n, (m%SPAT)] (op)= act( sum_k X[(m/SPAT)*K*SPAT + k*SPAT + m%SPAT]
//                                       * W[n*ldw + k] + bias[n] )
// One wave computes NT consecutive 16x16 output tiles sharing one A fragment
// per 32-deep K step (NT v_wmma per step).  A-row indices are clamped (not
// guarded): out-of-range rows read valid garbage whose products are discarded
// by the gm<M store mask, so the hot loop has no predication.  B rows and
// SPAT==1 A rows are loaded as float4 -> global_load_b128.
// Fragment packing follows ISA 7.12.2 (16-bit A 16x32, B 32x16, f32 C/D).
// ---------------------------------------------------------------------------
template <int SPAT, int NT, bool TAIL>
__global__ void __launch_bounds__(128)
gemm_kernel(const float* __restrict__ X, const float* __restrict__ W,
            const float* __restrict__ bias, float* __restrict__ Y,
            int M, int N, int K, int ldw, int act, int beta)
{
  const int lane = threadIdx.x & 31;
  const int wave = threadIdx.x >> 5;
  const int w0   = (blockIdx.y * (blockDim.x >> 5) + wave) * NT;  // first n-tile
  if (w0 * 16 >= N) return;               // wave-uniform exit, EXEC stays full
  const int mTile = blockIdx.x;
  const int half  = lane >> 4;
  const int l15   = lane & 15;

  const int mi  = mTile * 16 + l15;
  const int mic = (mi < M) ? mi : (M - 1);          // clamp, no per-load guard
  const int mq  = mic / SPAT;
  const int mr  = mic - mq * SPAT;
  const float* xrow = X + (size_t)mq * K * SPAT + mr;

  int ncol[NT];
  const float* wrow[NT];
#pragma unroll
  for (int t = 0; t < NT; ++t) {
    ncol[t] = (w0 + t) * 16 + l15;
    wrow[t] = W + (size_t)ncol[t] * ldw;
  }

  v8f zero = {};
  v8f acc[NT];
#pragma unroll
  for (int t = 0; t < NT; ++t) acc[t] = zero;

  const int Kfull = TAIL ? (K & ~31) : K;
  int k0 = 0;
  for (; k0 < Kfull; k0 += 32) {          // branchless hot loop
    v16h af;
    if (SPAT == 1) {
      // A rows contiguous: two 8-float runs at 32B-aligned offsets -> 4x b128
      const f4* xp0 = (const f4*)(xrow + k0 + (half << 3));
      const f4* xp1 = (const f4*)(xrow + k0 + 16 + (half << 3));
      f4 a0 = xp0[0], a1 = xp0[1], a2 = xp1[0], a3 = xp1[1];
#pragma unroll
      for (int q = 0; q < 4; ++q) {
        af[q]      = (_Float16)a0[q];
        af[4 + q]  = (_Float16)a1[q];
        af[8 + q]  = (_Float16)a2[q];
        af[12 + q] = (_Float16)a3[q];
      }
    } else {
      // strided TensLinear rows: scalar (still branchless) loads
#pragma unroll
      for (int e = 0; e < 16; ++e) {
        const int pr = e >> 1;
        const int ka = k0 + ((pr >> 2) << 4) + (half << 3) + ((pr & 3) << 1) + (e & 1);
        af[e] = (_Float16)xrow[(size_t)ka * SPAT];
      }
    }
#pragma unroll
    for (int t = 0; t < NT; ++t) {
      // B row: two 8-float runs at 32B-aligned offsets -> 4x b128
      const f4* wp = (const f4*)(wrow[t] + k0 + (half << 3));
      v16h bf;
#pragma unroll
      for (int q = 0; q < 4; ++q) {
        f4 wv = wp[q + ((q >> 1) << 1)];
        bf[q * 4 + 0] = (_Float16)wv[0];
        bf[q * 4 + 1] = (_Float16)wv[1];
        bf[q * 4 + 2] = (_Float16)wv[2];
        bf[q * 4 + 3] = (_Float16)wv[3];
      }
      acc[t] = __builtin_amdgcn_wmma_f32_16x16x32_f16(false, af, false, bf,
                                                      (short)0, acc[t],
                                                      false, false);
    }
  }
  if (TAIL && k0 < K) {                   // K%32==16 tail (Vout: K=48 only)
    v16h af;
#pragma unroll
    for (int e = 0; e < 16; ++e) {
      const int pr = e >> 1;
      const int ka = k0 + ((pr >> 2) << 4) + (half << 3) + ((pr & 3) << 1) + (e & 1);
      af[e] = (ka < K) ? (_Float16)xrow[(size_t)ka * SPAT] : (_Float16)0.0f;
    }
#pragma unroll
    for (int t = 0; t < NT; ++t) {
      v16h bf;
#pragma unroll
      for (int e = 0; e < 16; ++e) {
        const int kb = k0 + ((e >> 3) << 4) + (half << 3) + (e & 7);
        bf[e] = (kb < K) ? (_Float16)wrow[t][kb] : (_Float16)0.0f;
      }
      acc[t] = __builtin_amdgcn_wmma_f32_16x16x32_f16(false, af, false, bf,
                                                      (short)0, acc[t],
                                                      false, false);
    }
  }

#pragma unroll
  for (int t = 0; t < NT; ++t) {
    const float bv = bias ? bias[ncol[t]] : 0.0f;
#pragma unroll
    for (int r = 0; r < 8; ++r) {         // D layout: row = r + 8*half
      const int gm = mTile * 16 + r + (half << 3);
      if (gm < M) {
        float v = acc[t][r] + bv;
        if (act == 1) v = (v > 0.0f) ? v : 0.1f * v;   // LeakyReLU(0.1)
        const int gq = gm / SPAT;
        const int gr = gm - gq * SPAT;
        float* yp = Y + (size_t)gq * N * SPAT + (size_t)ncol[t] * SPAT + gr;
        if (beta) *yp += v; else *yp = v;
      }
    }
  }
}

// ---------------------------------------------------------------------------
// Edge preprocessing: 8 Gaussian radial features, lin_r (8->64, too small for
// WMMA -> scalar), and rh = tens_sigmoid(r * 7/r0).
// ---------------------------------------------------------------------------
__global__ void edge_prep_kernel(const float* __restrict__ rij,
                                 const float* __restrict__ Wl,
                                 const float* __restrict__ bl,
                                 float* __restrict__ rad, float* __restrict__ rh,
                                 int base, int cnt, float r0)
{
  int i = blockIdx.x * blockDim.x + threadIdx.x;
  if (i >= cnt) return;
  size_t gi = (size_t)(base + i);
  float x = rij[gi * 3 + 0], y = rij[gi * 3 + 1], z = rij[gi * 3 + 2];
  float d = sqrtf(x * x + y * y + z * z + 1e-12f);
  float feats[8];
  float w = r0 * 0.125f;
#pragma unroll
  for (int j = 0; j < 8; ++j) {
    float c = r0 * ((float)j / 7.0f);     // linspace(0, r0, 8)
    float u = (d - c) / w;
    feats[j] = expf(-0.5f * u * u);
  }
  float* ro = rad + (size_t)i * 64;
  for (int o = 0; o < 64; ++o) {
    float s = bl[o];
#pragma unroll
    for (int j = 0; j < 8; ++j) s += feats[j] * Wl[o * 8 + j];
    ro[o] = s;
  }
  float sc = 7.0f / r0;
  float sx = x * sc, sy = y * sc, sz = z * sc;
  float n = sqrtf(sx * sx + sy * sy + sz * sz + 1e-12f);
  float g = (2.0f / (1.0f + expf(-n)) - 1.0f) / n;
  rh[(size_t)i * 3 + 0] = sx * g;
  rh[(size_t)i * 3 + 1] = sy * g;
  rh[(size_t)i * 3 + 2] = sz * g;
}

// ---------------------------------------------------------------------------
// Edge scatter: per (edge-instance, channel) add rad, rad*rh, rad*rh(x)rh
// into per-slice node accumulators (40 MB total -> resident in 192 MB L2).
// ---------------------------------------------------------------------------
__global__ void edge_scatter_kernel(const float* __restrict__ rad,
                                    const float* __restrict__ rh,
                                    const int* __restrict__ src,
                                    float* __restrict__ acc_a,
                                    float* __restrict__ acc_v,
                                    float* __restrict__ acc_d,
                                    int base, int cnt, int E, int Nn)
{
  int t = blockIdx.x * blockDim.x + threadIdx.x;
  if (t >= cnt * 64) return;
  int i = t >> 6;
  int c = t & 63;
  int gi = base + i;
  int lb = gi / E;
  int e  = gi - lb * E;
  int sn = src[e];
  float val = rad[(size_t)i * 64 + c];
  float r0 = rh[(size_t)i * 3 + 0];
  float r1 = rh[(size_t)i * 3 + 1];
  float r2 = rh[(size_t)i * 3 + 2];
  size_t na = ((size_t)lb * Nn + sn) * 64 + c;
  atomicAdd(&acc_a[na], val);
  float* pv = acc_v + na * 3;
  atomicAdd(&pv[0], val * r0);
  atomicAdd(&pv[1], val * r1);
  atomicAdd(&pv[2], val * r2);
  float* pd = acc_d + na * 9;
  float rr[3] = {r0, r1, r2};
#pragma unroll
  for (int xx = 0; xx < 3; ++xx)
#pragma unroll
    for (int yy = 0; yy < 3; ++yy)
      atomicAdd(&pd[xx * 3 + yy], val * rr[xx] * rr[yy]);
}

// ---------------------------------------------------------------------------
// Pair products for the 13 tensor-product terms: Z[n,(l,i),s].  The GEMM with
// the P coupling tensor (shape (o,16,di), contiguous -> ldw=16*di) follows.
// ---------------------------------------------------------------------------
__global__ void pair_kernel(const float* __restrict__ U, const float* __restrict__ V,
                            float* __restrict__ Z, int n0, int nCh, int dI,
                            int sU, int sV, int sZ, int mode)
{
  int t = blockIdx.x * blockDim.x + threadIdx.x;
  int per = 16 * dI;
  if (t >= nCh * per) return;
  int nc  = t / per;
  int rem = t - nc * per;
  int l   = rem / dI;
  int i   = rem - l * dI;
  int node = n0 + nc;
  const float* u = U + ((size_t)node * 16 + l) * sU;
  const float* v = V + ((size_t)node * dI + i) * sV;
  float* z = Z + ((size_t)nc * per + rem) * sZ;
  switch (mode) {
    case 0:  z[0] = u[0] * v[0]; break;                                   // P000
    case 1:  z[0] = u[0]*v[0] + u[1]*v[1] + u[2]*v[2]; break;             // P110
    case 2:  { float s = 0.f; for (int k = 0; k < 9; ++k) s += u[k]*v[k];
               z[0] = s; } break;                                          // P220
    case 3:  { float a = u[0]; z[0]=a*v[0]; z[1]=a*v[1]; z[2]=a*v[2]; } break; // P011
    case 4:  { float a = v[0]; z[0]=u[0]*a; z[1]=u[1]*a; z[2]=u[2]*a; } break; // P101
    case 5:  for (int y2 = 0; y2 < 3; ++y2)                                // P121
               z[y2] = u[0]*v[y2] + u[1]*v[3+y2] + u[2]*v[6+y2];
             break;
    case 6:  for (int x2 = 0; x2 < 3; ++x2)                                // P211
               z[x2] = u[x2*3]*v[0] + u[x2*3+1]*v[1] + u[x2*3+2]*v[2];
             break;
    case 7:  z[0] = u[1]*v[2] - u[2]*v[1];                                 // P111 cross
             z[1] = u[2]*v[0] - u[0]*v[2];
             z[2] = u[0]*v[1] - u[1]*v[0];
             break;
    case 8:  { float a = u[0]; for (int k = 0; k < 9; ++k) z[k] = a*v[k]; } break; // P022
    case 9:  for (int x2 = 0; x2 < 3; ++x2)                                // P112 outer
               for (int y2 = 0; y2 < 3; ++y2) z[x2*3+y2] = u[x2]*v[y2];
             break;
    case 10: { float a = v[0]; for (int k = 0; k < 9; ++k) z[k] = u[k]*a; } break; // P202
    case 11: for (int x2 = 0; x2 < 3; ++x2)                                // P222 matmul
               for (int y2 = 0; y2 < 3; ++y2)
                 z[x2*3+y2] = u[x2*3]*v[y2] + u[x2*3+1]*v[3+y2] + u[x2*3+2]*v[6+y2];
             break;
    default: // 12: P212 — row-x of u crossed with v
      for (int x2 = 0; x2 < 3; ++x2) {
        const float* ur = u + x2*3;
        z[x2*3+0] = ur[1]*v[2] - ur[2]*v[1];
        z[x2*3+1] = ur[2]*v[0] - ur[0]*v[2];
        z[x2*3+2] = ur[0]*v[1] - ur[1]*v[0];
      }
      break;
  }
}

// ---------------------------------------------------------------------------
// time_embed: g[b,o] = b[o] + sum_k sin(t*f_k) W[o,k] + cos(t*f_k) W[o,64+k]
// ---------------------------------------------------------------------------
__global__ void time_embed_kernel(const float* __restrict__ t,
                                  const float* __restrict__ W,
                                  const float* __restrict__ b,
                                  float* __restrict__ g, int d)
{
  int bb = blockIdx.x, o = threadIdx.x;
  if (o >= d) return;
  float tv = t[bb];
  float s = b[o];
  for (int k = 0; k < 64; ++k) {
    float f = expf(-(float)k * (1.0f / 63.0f));
    float ph = tv * f;
    s += sinf(ph) * W[o * 128 + k] + cosf(ph) * W[o * 128 + 64 + k];
  }
  g[bb * d + o] = s;
}

__global__ void scale_mod_kernel(float* __restrict__ Xb, const float* __restrict__ g,
                                 int total, int Cspat, int Nn, int C, int spat)
{
  int i = blockIdx.x * blockDim.x + threadIdx.x;
  if (i >= total) return;
  int bn = i / Cspat;
  int c  = (i - bn * Cspat) / spat;
  Xb[i] *= g[(bn / Nn) * C + c];
}

__global__ void axpy_kernel(float* __restrict__ dst, const float* __restrict__ src, int n)
{
  int i = blockIdx.x * blockDim.x + threadIdx.x;
  if (i < n) dst[i] += src[i];
}

__global__ void add2_kernel(float* __restrict__ dst, const float* __restrict__ a,
                            const float* __restrict__ b, int n)
{
  int i = blockIdx.x * blockDim.x + threadIdx.x;
  if (i < n) dst[i] = a[i] + b[i];
}

// tens_sigmoid over 3-vectors (per (node,channel))
__global__ void tsig3_kernel(float* __restrict__ Xb, int rows)
{
  int i = blockIdx.x * blockDim.x + threadIdx.x;
  if (i >= rows) return;
  float* p = Xb + (size_t)i * 3;
  float n = sqrtf(p[0]*p[0] + p[1]*p[1] + p[2]*p[2] + 1e-12f);
  float s = (2.0f / (1.0f + expf(-n)) - 1.0f) / n;
  p[0] *= s; p[1] *= s; p[2] *= s;
}

// TensGroupNorm (8 groups); subMean only for inds==0
__global__ void tgn_kernel(float* __restrict__ Xb, const float* __restrict__ gamma,
                           int BNtot, int C, int spat, int subMean)
{
  int t = blockIdx.x * blockDim.x + threadIdx.x;
  if (t >= BNtot * 8) return;
  int bn = t >> 3, g = t & 7;
  int cpg = C >> 3;
  int cnt = cpg * spat;
  float* p = Xb + (size_t)bn * C * spat + (size_t)g * cnt;
  float mean = 0.0f;
  if (subMean) {
    for (int i = 0; i < cnt; ++i) mean += p[i];
    mean /= (float)cnt;
  }
  float ms = 0.0f;
  for (int i = 0; i < cnt; ++i) { float v = p[i] - mean; ms += v * v; }
  ms /= (float)cnt;
  float inv = 1.0f / sqrtf(ms + 1e-5f);
  for (int i = 0; i < cnt; ++i) {
    int c = g * cpg + i / spat;
    p[i] = (p[i] - mean) * inv * gamma[c];
  }
}

// out[m,x,y] = xd[m,x,y] + Td[m,y,x]
__global__ void trans_add_kernel(float* __restrict__ out, const float* __restrict__ xd,
                                 const float* __restrict__ Td, int nMat)
{
  int t = blockIdx.x * blockDim.x + threadIdx.x;
  if (t >= nMat * 9) return;
  int m = t / 9, s = t - m * 9;
  int x = s / 3, y = s - x * 3;
  out[t] = xd[t] + Td[(size_t)m * 9 + y * 3 + x];
}

// ============================================================================
// Host orchestration
// ============================================================================
static void launch_gemm(hipStream_t stream, const float* X, const float* W,
                        const float* bias, float* Y, int M, int N, int K,
                        int ldw, int spat, int act, int beta)
{
  int tilesN = N / 16;
  int nt = (tilesN % 4 == 0) ? 4 : (tilesN % 3 == 0) ? 3
         : (tilesN % 2 == 0) ? 2 : 1;
  int wavesNeeded = tilesN / nt;
  int wpb = wavesNeeded < 4 ? wavesNeeded : 4;
  dim3 grid(cdiv(M, 16), cdiv(wavesNeeded, wpb));
  dim3 block(32 * wpb);
  bool tail = (K & 31) != 0;
#define GCASE(S, T, TL) gemm_kernel<S, T, TL><<<grid, block, 0, stream>>>(X, W, bias, Y, M, N, K, ldw, act, beta); break
#define GSWITCH(TL)                                                            \
  switch (spat * 8 + nt) {                                                     \
    case 1 * 8 + 1: GCASE(1, 1, TL);                                           \
    case 1 * 8 + 2: GCASE(1, 2, TL);                                           \
    case 1 * 8 + 3: GCASE(1, 3, TL);                                           \
    case 1 * 8 + 4: GCASE(1, 4, TL);                                           \
    case 3 * 8 + 1: GCASE(3, 1, TL);                                           \
    case 3 * 8 + 2: GCASE(3, 2, TL);                                           \
    case 3 * 8 + 3: GCASE(3, 3, TL);                                           \
    case 3 * 8 + 4: GCASE(3, 4, TL);                                           \
    case 9 * 8 + 1: GCASE(9, 1, TL);                                           \
    case 9 * 8 + 2: GCASE(9, 2, TL);                                           \
    case 9 * 8 + 3: GCASE(9, 3, TL);                                           \
    default:        GCASE(9, 4, TL);                                           \
  }
  if (tail) { GSWITCH(true) } else { GSWITCH(false) }
#undef GSWITCH
#undef GCASE
}

extern "C" void kernel_launch(void* const* d_in, const int* in_sizes, int n_in,
                              void* d_out, int out_size, void* d_ws, size_t ws_size,
                              hipStream_t stream)
{
  (void)in_sizes; (void)out_size;
  if (n_in < 94) return;                       // expect 8 inputs + 86 param leaves

  constexpr int Bz = 4, NN = 1000, BN = 4000, E = 16000;
  constexpr int ECH = 32000;                   // edge-instance chunk
  constexpr int NCH = 500;                     // node chunk for P-terms

  const float* in_xa = (const float*)d_in[0];
  const float* in_xv = (const float*)d_in[1];
  const float* in_xd = (const float*)d_in[2];
  const float* t_in  = (const float*)d_in[3];
  const float* rij0  = (const float*)d_in[4];
  const float* rij1  = (const float*)d_in[5];
  const int*   src0  = (const int*)d_in[6];
  const int*   src1  = (const int*)d_in[7];
#define PRM(i) ((const float*)d_in[(i)])
  // params flattened jax-pytree style (sorted keys), appended after inputs:
  //  8 Ra | 9 Rd | 10 Rv
  // 11-22  dt0: Wa Wd Wv linr.W linr.b mlp.Wd l1W l1b l2W l2b l3W l3b
  // 23-34  dt1: (same)
  // 35 Wdt | 36-42 mlp_a: Wd l1W l1b l2W l2b l3W l3b | 43-46 mlp_v: W1 W2 W3 Wd
  // 47-50 tlm_a: Wi Wo embW embb | 51-54 tlm_d | 55-58 tlm_v
  // 59 Ain 60 Aout 61 Din 62 Dout
  // 63-75  L000 L011 L022 L101 L110 L111 L112 L121 L202 L211 L212 L220 L222
  // 76-88  P000 P011 P022 P101 P110 P111 P112 P121 P202 P211 P212 P220 P222
  // 89 Vin 90 Vout 91 ga 92 gd 93 gv

  // ---------------- workspace layout (floats) ----------------
  float* w = (float*)d_ws;
  size_t cur = 0;
  auto alloc = [&](size_t n) { float* p = w + cur; cur += n; return p; };

  float* acc_a0 = alloc(256000);   float* acc_v0 = alloc(768000);
  float* acc_d0 = alloc(2304000);
  float* acc_a1 = alloc(512000);   float* acc_v1 = alloc(1536000);
  float* acc_d1 = alloc(4608000);                     // acc block: 9,984,000
  float* xa = alloc(512000); float* xv = alloc(768000); float* xd = alloc(1152000);
  float* l000b = alloc(64000);  float* l011b = alloc(64000);  float* l022b = alloc(64000);
  float* l110b = alloc(192000); float* l101b = alloc(192000); float* l121b = alloc(192000);
  float* l111b = alloc(192000); float* l112b = alloc(192000);
  float* l220b = alloc(576000); float* l211b = alloc(576000); float* l202b = alloc(576000);
  float* l222b = alloc(576000); float* l212b = alloc(576000);
  float* amb = alloc(256000); float* vmb = alloc(576000); float* dmb = alloc(1152000);
  float* yab = alloc(256000); float* yvb = alloc(576000); float* ydb = alloc(1152000);
  float* scr = alloc(14600000);                       // shared edge/node scratch
  if (ws_size < cur * sizeof(float)) return;          // not enough scratch

  // edge-stage view of scr
  float* e_rad = scr;
  float* e_h1  = e_rad + (size_t)ECH * 64;
  float* e_h2  = e_h1  + (size_t)ECH * 128;
  float* e_out = e_h2  + (size_t)ECH * 128;
  float* e_rh  = e_out + (size_t)ECH * 64;
  // node-stage view of scr (edge buffers dead by then)
  float* Zbuf = scr;
  float* Ta   = Zbuf + 4608000;
  float* Tv   = Ta + 512000;
  float* Td   = Tv + 768000;
  float* h1n  = Td + 1152000;
  float* h2n  = h1n + 1024000;
  float* h1v  = h2n + 1024000;
  float* h2v  = h1v + 1536000;
  float* T1a  = h2v + 1536000;
  float* T1v  = T1a + 512000;
  float* T1d  = T1v + 768000;
  float* g_a  = T1d + 1152000;
  float* g_v  = g_a + 512;
  float* g_d  = g_v + 256;

  auto gemm = [&](const float* X, const float* Wt, const float* bias, float* Y,
                  int M, int Nt, int K, int ldw, int spat, int act, int beta) {
    launch_gemm(stream, X, Wt, bias, Y, M, Nt, K, ldw, spat, act, beta);
  };

  // ================= Stage A: edge branches ==========================
  hipMemsetAsync(acc_a0, 0, 9984000 * sizeof(float), stream);

  auto run_dt = [&](const float* rij, const int* src, int LB, int pbase,
                    float* aa, float* av, float* ad) {
    const float* linrW = PRM(pbase + 3); const float* linrb = PRM(pbase + 4);
    const float* mWd = PRM(pbase + 5);
    const float* l1W = PRM(pbase + 6); const float* l1b = PRM(pbase + 7);
    const float* l2W = PRM(pbase + 8); const float* l2b = PRM(pbase + 9);
    const float* l3W = PRM(pbase + 10); const float* l3b = PRM(pbase + 11);
    int total = LB * E;
    for (int base = 0; base < total; base += ECH) {
      int cnt = total - base; if (cnt > ECH) cnt = ECH;
      edge_prep_kernel<<<cdiv(cnt, 256), 256, 0, stream>>>(rij, linrW, linrb,
                                                           e_rad, e_rh, base, cnt, 6.0f);
      gemm(e_rad, l1W, l1b, e_h1, cnt, 128, 64, 64, 1, 1, 0);
      gemm(e_h1, l2W, l2b, e_h2, cnt, 128, 128, 128, 1, 1, 0);
      gemm(e_h2, l3W, l3b, e_out, cnt, 64, 128, 128, 1, 0, 0);
      gemm(e_rad, mWd, nullptr, e_out, cnt, 64, 64, 64, 1, 0, 1);
      edge_scatter_kernel<<<cdiv(cnt * 64, 256), 256, 0, stream>>>(
          e_out, e_rh, src, aa, av, ad, base, cnt, E, NN);
    }
  };
  run_dt(rij0, src0, Bz,     11, acc_a0, acc_v0, acc_d0);   // dt0, L0=1
  run_dt(rij1, src1, 2 * Bz, 23, acc_a1, acc_v1, acc_d1);   // dt1, L1=2

  // ================= Stage B: residual proj + fuse displacement ======
  gemm(in_xa, PRM(8), nullptr, xa, BN, 128, 128, 128, 1, 0, 0);          // Ra
  gemm(acc_a0, PRM(11), nullptr, xa, BN, 128, 64, 64, 1, 0, 1);          // dt0.Wa
  for (int l = 0; l < 2; ++l)
    gemm(acc_a1 + (size_t)l * 256000, PRM(23) + l * 64, nullptr, xa,
         BN, 128, 64, 128, 1, 0, 1);                                     // dt1.Wa
  gemm(in_xv, PRM(10), nullptr, xv, BN * 3, 64, 64, 64, 3, 0, 0);        // Rv
  gemm(acc_v0, PRM(13), nullptr, xv, BN * 3, 64, 64, 64, 3, 0, 1);       // dt0.Wv
  for (int l = 0; l < 2; ++l)
    gemm(acc_v1 + (size_t)l * 768000, PRM(25) + l * 64, nullptr, xv,
         BN * 3, 64, 64, 128, 3, 0, 1);                                  // dt1.Wv
  gemm(in_xd, PRM(9), nullptr, xd, BN * 9, 32, 32, 32, 9, 0, 0);         // Rd
  gemm(acc_d0, PRM(12), nullptr, xd, BN * 9, 32, 64, 64, 9, 0, 1);       // dt0.Wd
  for (int l = 0; l < 2; ++l)
    gemm(acc_d1 + (size_t)l * 2304000, PRM(24) + l * 64, nullptr, xd,
         BN * 9, 32, 64, 128, 9, 0, 1);                                  // dt1.Wd

  // ================= Stage C: self tensor products ===================
  gemm(xa, PRM(63), nullptr, l000b, BN, 16, 128, 128, 1, 0, 0);
  gemm(xa, PRM(64), nullptr, l011b, BN, 16, 128, 128, 1, 0, 0);
  gemm(xa, PRM(65), nullptr, l022b, BN, 16, 128, 128, 1, 0, 0);
  gemm(xa, PRM(59), nullptr, amb,   BN, 64, 128, 128, 1, 0, 0);          // Ain
  gemm(xv, PRM(67), nullptr, l110b, BN * 3, 16, 64, 64, 3, 0, 0);
  gemm(xv, PRM(66), nullptr, l101b, BN * 3, 16, 64, 64, 3, 0, 0);
  gemm(xv, PRM(70), nullptr, l121b, BN * 3, 16, 64, 64, 3, 0, 0);
  gemm(xv, PRM(68), nullptr, l111b, BN * 3, 16, 64, 64, 3, 0, 0);
  gemm(xv, PRM(69), nullptr, l112b, BN * 3, 16, 64, 64, 3, 0, 0);
  gemm(xv, PRM(89), nullptr, vmb,   BN * 3, 48, 64, 64, 3, 0, 0);        // Vin
  gemm(xd, PRM(74), nullptr, l220b, BN * 9, 16, 32, 32, 9, 0, 0);
  gemm(xd, PRM(72), nullptr, l211b, BN * 9, 16, 32, 32, 9, 0, 0);
  gemm(xd, PRM(71), nullptr, l202b, BN * 9, 16, 32, 32, 9, 0, 0);
  gemm(xd, PRM(75), nullptr, l222b, BN * 9, 16, 32, 32, 9, 0, 0);
  gemm(xd, PRM(73), nullptr, l212b, BN * 9, 16, 32, 32, 9, 0, 0);
  gemm(xd, PRM(61), nullptr, dmb,   BN * 9, 32, 32, 32, 9, 0, 0);        // Din

  struct PT { const float* P; const float* U; const float* V; float* Yb;
              int dI, No, mode, sU, sV, sZ, beta; };
  const PT terms[13] = {
    { PRM(76), l000b, amb, yab, 64, 64, 0, 1, 1, 1, 0 },  // P000
    { PRM(80), l110b, vmb, yab, 48, 64, 1, 3, 3, 1, 1 },  // P110
    { PRM(87), l220b, dmb, yab, 32, 64, 2, 9, 9, 1, 1 },  // P220
    { PRM(77), l011b, vmb, yvb, 48, 48, 3, 1, 3, 3, 0 },  // P011
    { PRM(79), l101b, amb, yvb, 64, 48, 4, 3, 1, 3, 1 },  // P101
    { PRM(83), l121b, dmb, yvb, 32, 48, 5, 3, 9, 3, 1 },  // P121
    { PRM(85), l211b, vmb, yvb, 48, 48, 6, 9, 3, 3, 1 },  // P211
    { PRM(81), l111b, vmb, yvb, 48, 48, 7, 3, 3, 3, 1 },  // P111
    { PRM(78), l022b, dmb, ydb, 32, 32, 8, 1, 9, 9, 0 },  // P022
    { PRM(82), l112b, vmb, ydb, 48, 32, 9, 3, 3, 9, 1 },  // P112
    { PRM(84), l202b, amb, ydb, 64, 32, 10, 9, 1, 9, 1 }, // P202
    { PRM(88), l222b, dmb, ydb, 32, 32, 11, 9, 9, 9, 1 }, // P222
    { PRM(86), l212b, vmb, ydb, 48, 32, 12, 9, 3, 9, 1 }, // P212
  };
  for (int ti = 0; ti < 13; ++ti) {
    const PT& tm = terms[ti];
    for (int n0 = 0; n0 < BN; n0 += NCH) {
      pair_kernel<<<cdiv(NCH * 16 * tm.dI, 256), 256, 0, stream>>>(
          tm.U, tm.V, Zbuf, n0, NCH, tm.dI, tm.sU, tm.sV, tm.sZ, tm.mode);
      gemm(Zbuf, tm.P, nullptr, tm.Yb + (size_t)n0 * tm.No * tm.sZ,
           NCH * tm.sZ, tm.No, 16 * tm.dI, 16 * tm.dI, tm.sZ, 0, tm.beta);
    }
  }
  // out-proj + residual + group norm
  gemm(yab, PRM(60), nullptr, Ta, BN, 128, 64, 64, 1, 0, 0);             // Aout
  axpy_kernel<<<cdiv(512000, 256), 256, 0, stream>>>(xa, Ta, 512000);
  tgn_kernel<<<cdiv(BN * 8, 256), 256, 0, stream>>>(xa, PRM(91), BN, 128, 1, 1);
  gemm(yvb, PRM(90), nullptr, Tv, BN * 3, 64, 48, 48, 3, 0, 0);          // Vout (K=48 tail)
  axpy_kernel<<<cdiv(768000, 256), 256, 0, stream>>>(xv, Tv, 768000);
  tgn_kernel<<<cdiv(BN * 8, 256), 256, 0, stream>>>(xv, PRM(93), BN, 64, 3, 0);
  gemm(ydb, PRM(62), nullptr, Td, BN * 9, 32, 32, 32, 9, 0, 0);          // Dout
  axpy_kernel<<<cdiv(1152000, 256), 256, 0, stream>>>(xd, Td, 1152000);
  tgn_kernel<<<cdiv(BN * 8, 256), 256, 0, stream>>>(xd, PRM(92), BN, 32, 9, 0);

  // ================= Stage D: local MLPs =============================
  time_embed_kernel<<<Bz, 128, 0, stream>>>(t_in, PRM(49), PRM(50), g_a, 128);
  time_embed_kernel<<<Bz, 64,  0, stream>>>(t_in, PRM(57), PRM(58), g_v, 64);
  time_embed_kernel<<<Bz, 32,  0, stream>>>(t_in, PRM(53), PRM(54), g_d, 32);

  gemm(xa, PRM(47), nullptr, T1a, BN, 128, 128, 128, 1, 0, 0);           // tlm_a.Wi
  scale_mod_kernel<<<cdiv(512000, 256), 256, 0, stream>>>(T1a, g_a, 512000, 128, NN, 128, 1);
  gemm(T1a, PRM(48), nullptr, xa, BN, 128, 128, 128, 1, 0, 1);           // += Wo

  gemm(xv, PRM(55), nullptr, T1v, BN * 3, 64, 64, 64, 3, 0, 0);          // tlm_v.Wi
  scale_mod_kernel<<<cdiv(768000, 256), 256, 0, stream>>>(T1v, g_v, 768000, 192, NN, 64, 3);
  gemm(T1v, PRM(56), nullptr, xv, BN * 3, 64, 64, 64, 3, 0, 1);

  gemm(xd, PRM(51), nullptr, T1d, BN * 9, 32, 32, 32, 9, 0, 0);          // tlm_d.Wi
  scale_mod_kernel<<<cdiv(1152000, 256), 256, 0, stream>>>(T1d, g_d, 1152000, 288, NN, 32, 9);
  gemm(T1d, PRM(52), nullptr, xd, BN * 9, 32, 32, 32, 9, 0, 1);

  float* out_xa = (float*)d_out;
  float* out_xv = out_xa + 512000;
  float* out_xd = out_xa + 1280000;

  // mlp_a
  gemm(xa, PRM(37), PRM(38), h1n, BN, 256, 128, 128, 1, 1, 0);
  gemm(h1n, PRM(39), PRM(40), h2n, BN, 256, 256, 256, 1, 1, 0);
  gemm(h2n, PRM(41), PRM(42), Ta, BN, 128, 256, 256, 1, 0, 0);
  gemm(xa, PRM(36), nullptr, Ta, BN, 128, 128, 128, 1, 0, 1);            // + x@Wd
  add2_kernel<<<cdiv(512000, 256), 256, 0, stream>>>(out_xa, xa, Ta, 512000);

  // mlp_v (VectorMLP with tens_sigmoid)
  gemm(xv, PRM(43), nullptr, h1v, BN * 3, 128, 64, 64, 3, 0, 0);
  tsig3_kernel<<<cdiv(BN * 128, 256), 256, 0, stream>>>(h1v, BN * 128);
  gemm(h1v, PRM(44), nullptr, h2v, BN * 3, 128, 128, 128, 3, 0, 0);
  tsig3_kernel<<<cdiv(BN * 128, 256), 256, 0, stream>>>(h2v, BN * 128);
  gemm(h2v, PRM(45), nullptr, Tv, BN * 3, 64, 128, 128, 3, 0, 0);
  gemm(xv, PRM(46), nullptr, Tv, BN * 3, 64, 64, 64, 3, 0, 1);
  add2_kernel<<<cdiv(768000, 256), 256, 0, stream>>>(out_xv, xv, Tv, 768000);

  // x_d tail: x_d + (Wdt x_d)^T
  gemm(xd, PRM(35), nullptr, Td, BN * 9, 32, 32, 32, 9, 0, 0);
  trans_add_kernel<<<cdiv(1152000, 256), 256, 0, stream>>>(out_xd, xd, Td, BN * 32);
#undef PRM
}
